// GCNModel_22402549416514
// MI455X (gfx1250) — matrix operand order repaired and checked by reading, not compiled, not run on hardware
//
#include <hip/hip_runtime.h>

#define NUSER 30000
#define NITEM 40000
#define NTOT  70000
#define DDIM  64
#define IMGF  4096
#define TXTF  768
#define EPSBN 1e-5f
#define SLOPE 0.01f

typedef __bf16 bf16_t;
typedef bf16_t v16bf __attribute__((ext_vector_type(16)));
typedef float  v8f   __attribute__((ext_vector_type(8)));
typedef int    v4i   __attribute__((ext_vector_type(4)));

#if defined(__has_builtin)
#if __has_builtin(__builtin_amdgcn_global_load_async_to_lds_b128) && \
    __has_builtin(__builtin_amdgcn_s_wait_asynccnt)
#define HAVE_ASYNC_LDS 1
#endif
#endif

#ifdef HAVE_ASYNC_LDS
typedef __attribute__((address_space(1))) v4i* gv4i_p;  // global v4i32*
typedef __attribute__((address_space(3))) v4i* sv4i_p;  // LDS v4i32*
#endif

__device__ __forceinline__ float leakyf(float x) { return x >= 0.f ? x : SLOPE * x; }

// copy one 4KB B-chunk (32 rows x 64 bf16) global -> LDS; 128 threads x 32B
__device__ __forceinline__ void copy_b_chunk(bf16_t* dst, const bf16_t* src, int tid) {
#ifdef HAVE_ASYNC_LDS
  gv4i_p g = (gv4i_p)(src + tid * 16);
  sv4i_p s = (sv4i_p)(dst + tid * 16);
  __builtin_amdgcn_global_load_async_to_lds_b128(g, s, 0, 0);
  __builtin_amdgcn_global_load_async_to_lds_b128(g, s, 16, 0);
#else
  ((float4*)dst)[tid * 2 + 0] = ((const float4*)src)[tid * 2 + 0];
  ((float4*)dst)[tid * 2 + 1] = ((const float4*)src)[tid * 2 + 1];
#endif
}

__device__ __forceinline__ void wait_async_copies() {
#ifdef HAVE_ASYNC_LDS
  __builtin_amdgcn_s_wait_asynccnt(0);
#endif
}

// ---------------- elementwise / prep kernels ----------------

__global__ void k_concat(const float* __restrict__ ue, const float* __restrict__ ie,
                         float* __restrict__ e0) {
  int i = blockIdx.x * blockDim.x + threadIdx.x;
  if (i >= NTOT * DDIM) return;
  int split = NUSER * DDIM;
  e0[i] = (i < split) ? ue[i] : ie[i - split];
}

__global__ void k_content(float* __restrict__ e0, const float* __restrict__ e1,
                          const float* __restrict__ e2) {
  int i = blockIdx.x * blockDim.x + threadIdx.x;
  if (i >= NTOT * DDIM) return;
  e0[i] = (e0[i] + e1[i] + e2[i]) * (1.f / 3.f);
}

// W: [64, K] row-major (fp32)  ->  Wt: [K, 64] bf16 (so each 32-row K chunk is a contiguous 4KB)
__global__ void k_wt(const float* __restrict__ W, bf16_t* __restrict__ Wt, int K) {
  int i = blockIdx.x * blockDim.x + threadIdx.x;
  if (i >= K * DDIM) return;
  int k = i >> 6, n = i & 63;
  Wt[i] = (bf16_t)W[n * K + k];
}

// ---------------- COO SpMM: y[row] += val * x[col], 16 lanes/edge ----------------
// Edge lists are streamed once -> non-temporal so they don't evict the L2-resident
// dense matrices (x/y are ~18MB each; L2 is 192MB).

__global__ void k_spmm(const int* __restrict__ rows, const int* __restrict__ cols,
                       const float* __restrict__ vals, const float* __restrict__ x,
                       float* __restrict__ y, int nnz) {
  int t = blockIdx.x * blockDim.x + threadIdx.x;
  int e = t >> 4;
  if (e >= nnz) return;
  int l = t & 15;
  int r = __builtin_nontemporal_load(rows + e);
  int c = __builtin_nontemporal_load(cols + e);
  float v = __builtin_nontemporal_load(vals + e);
  const float4 xv = *(const float4*)(x + (size_t)c * DDIM + l * 4);
  float* yp = y + (size_t)r * DDIM + l * 4;
  unsafeAtomicAdd(yp + 0, v * xv.x);
  unsafeAtomicAdd(yp + 1, v * xv.y);
  unsafeAtomicAdd(yp + 2, v * xv.z);
  unsafeAtomicAdd(yp + 3, v * xv.w);
}

// ---------------- WMMA GEMM: C[M,64] = A[M,K] (fp32) @ Bt[K,64] (bf16) ----------------
// One wave per 16-row M-tile, 4 N-tiles of 16 in v8f accumulators.
// B chunks (32x64 bf16 = 4KB) are staged into LDS with async copies, double-buffered:
// all 4 waves share the same B. A is loaded from global and converted f32->bf16 in-register.
// A layout (16-bit 16x32): lane L holds row M=L%16; elem e -> K = (e/8)*16 + (L/16)*8 + e%8.
// B layout (16-bit 32x16): lane L holds col N=L%16; elem e -> K = (e/8)*16 + (L/16)*8 + e%8.

__global__ void __launch_bounds__(128) k_gemm64(const float* __restrict__ A,
                                                const bf16_t* __restrict__ Bt,
                                                float* __restrict__ C, int M, int K) {
  __shared__ bf16_t Bs[2][32 * DDIM];  // 8KB double buffer
  int tid = threadIdx.x;
  int lane = tid & 31;
  int wave = tid >> 5;
  int mt = blockIdx.x * 4 + wave;
  int m0 = mt * 16;
  if (m0 > M - 16) m0 = M - 16;  // ragged tail: duplicate tile, byte-identical stores (benign)
  int mrow = m0 + (lane & 15);
  int h8 = (lane >> 4) * 8;
  const float* arow = A + (size_t)mrow * K;

  v8f acc0 = {0.f,0.f,0.f,0.f,0.f,0.f,0.f,0.f};
  v8f acc1 = acc0, acc2 = acc0, acc3 = acc0;

  int nk = K >> 5;  // K % 32 == 0 for all layers here

  copy_b_chunk(Bs[0], Bt, tid);
  wait_async_copies();
  __syncthreads();

  for (int kc = 0; kc < nk; ++kc) {
    int cur = kc & 1;
    if (kc + 1 < nk)  // prefetch next B chunk into the other buffer (async, overlaps WMMA)
      copy_b_chunk(Bs[cur ^ 1], Bt + (size_t)(kc + 1) * 32 * DDIM, tid);

    int k0 = kc * 32;
    float4 a0 = *(const float4*)(arow + k0 + h8);
    float4 a1 = *(const float4*)(arow + k0 + h8 + 4);
    float4 a2 = *(const float4*)(arow + k0 + 16 + h8);
    float4 a3 = *(const float4*)(arow + k0 + 16 + h8 + 4);
    if (kc + 1 < nk) __builtin_prefetch(arow + k0 + 32 + h8, 0, 0);

    v16bf av;
    av[0]=(bf16_t)a0.x;  av[1]=(bf16_t)a0.y;  av[2]=(bf16_t)a0.z;  av[3]=(bf16_t)a0.w;
    av[4]=(bf16_t)a1.x;  av[5]=(bf16_t)a1.y;  av[6]=(bf16_t)a1.z;  av[7]=(bf16_t)a1.w;
    av[8]=(bf16_t)a2.x;  av[9]=(bf16_t)a2.y;  av[10]=(bf16_t)a2.z; av[11]=(bf16_t)a2.w;
    av[12]=(bf16_t)a3.x; av[13]=(bf16_t)a3.y; av[14]=(bf16_t)a3.z; av[15]=(bf16_t)a3.w;

    int bn = lane & 15;
    v16bf b0, b1, b2, b3;
    for (int s = 0; s < 16; ++s) {
      int kk = (s < 8) ? (h8 + s) : (16 + h8 + s - 8);
      const bf16_t* bp = &Bs[cur][kk * DDIM + bn];
      b0[s] = bp[0]; b1[s] = bp[16]; b2[s] = bp[32]; b3[s] = bp[48];
    }

    acc0 = __builtin_amdgcn_wmma_f32_16x16x32_bf16(false, av, false, b0, (short)0, acc0, false, false);
    acc1 = __builtin_amdgcn_wmma_f32_16x16x32_bf16(false, av, false, b1, (short)0, acc1, false, false);
    acc2 = __builtin_amdgcn_wmma_f32_16x16x32_bf16(false, av, false, b2, (short)0, acc2, false, false);
    acc3 = __builtin_amdgcn_wmma_f32_16x16x32_bf16(false, av, false, b3, (short)0, acc3, false, false);

    wait_async_copies();  // next buffer landed
    __syncthreads();      // everyone done reading cur before it is overwritten next round
  }

  // C/D layout: lane L -> N = L%16 ; VGPR r -> M = (L/16)*8 + r
  int n = lane & 15;
  int mbase = m0 + (lane >> 4) * 8;
  for (int r = 0; r < 8; ++r) {
    float* crow = C + (size_t)(mbase + r) * DDIM + n;
    crow[0]  = acc0[r];
    crow[16] = acc1[r];
    crow[32] = acc2[r];
    crow[48] = acc3[r];
  }
}

// ---------------- BatchNorm (training-mode batch stats) ----------------

__global__ void __launch_bounds__(256) k_bnstats(const float* __restrict__ Y,
                                                 float* __restrict__ stats, int M) {
  __shared__ float s1[256], s2[256];
  int col = blockIdx.x;  // 64 blocks
  float sum = 0.f, sq = 0.f;
  for (int r = threadIdx.x; r < M; r += 256) {
    float v = Y[(size_t)r * DDIM + col];
    sum += v; sq += v * v;
  }
  s1[threadIdx.x] = sum; s2[threadIdx.x] = sq;
  __syncthreads();
  for (int s = 128; s > 0; s >>= 1) {
    if (threadIdx.x < s) {
      s1[threadIdx.x] += s1[threadIdx.x + s];
      s2[threadIdx.x] += s2[threadIdx.x + s];
    }
    __syncthreads();
  }
  if (threadIdx.x == 0) { stats[col] = s1[0]; stats[64 + col] = s2[0]; }
}

// out = leaky((Y - mean)*rsqrt(var+eps)*g + beta) [+ res]   (linear bias cancels in BN)
__global__ void k_bnapply(const float* __restrict__ Y, const float* __restrict__ stats,
                          const float* __restrict__ g, const float* __restrict__ beta,
                          const float* __restrict__ res, float* __restrict__ out, int M) {
  int i = blockIdx.x * blockDim.x + threadIdx.x;
  if (i >= M * DDIM) return;
  int c = i & 63;
  float inv_m = 1.f / (float)M;
  float mean = stats[c] * inv_m;
  float var = stats[64 + c] * inv_m - mean * mean;
  float v = (Y[i] - mean) * rsqrtf(var + EPSBN) * g[c] + beta[c];
  v = leakyf(v);
  if (res) v += res[i];
  out[i] = v;
}

// out = item_emb * sigmoid(Y + gb)
__global__ void k_gate(const float* __restrict__ Y, const float* __restrict__ gb,
                       const float* __restrict__ item, float* __restrict__ out) {
  int i = blockIdx.x * blockDim.x + threadIdx.x;
  if (i >= NITEM * DDIM) return;
  int c = i & 63;
  float s = 1.f / (1.f + __expf(-(Y[i] + gb[c])));
  out[i] = item[i] * s;
}

// s[row] = sum_j leaky(H[row,j] + b1[j]) * w2[j]
__global__ void k_score(const float* __restrict__ H, const float* __restrict__ b1,
                        const float* __restrict__ w2, float* __restrict__ s, int M) {
  int r = blockIdx.x * blockDim.x + threadIdx.x;
  if (r >= M) return;
  const float* hr = H + (size_t)r * DDIM;
  float acc = 0.f;
  for (int j = 0; j < DDIM; ++j) acc += leakyf(hr[j] + b1[j]) * w2[j];
  s[r] = acc;
}

// out = content + softmax(s_img,s_txt)[0]*img_ui + [1]*txt_ui
__global__ void k_fuse(const float* __restrict__ content, const float* __restrict__ iui,
                       const float* __restrict__ tui, const float* __restrict__ si,
                       const float* __restrict__ st, float* __restrict__ out) {
  int i = blockIdx.x * blockDim.x + threadIdx.x;
  if (i >= NTOT * DDIM) return;
  int r = i >> 6;
  float a = si[r], b = st[r];
  float m = fmaxf(a, b);
  float ea = __expf(a - m), eb = __expf(b - m);
  float w = ea / (ea + eb);
  out[i] = content[i] + w * iui[i] + (1.f - w) * tui[i];
}

// ---------------- host-side sequencing ----------------

static inline int cdiv(long a, long b) { return (int)((a + b - 1) / b); }

extern "C" void kernel_launch(void* const* d_in, const int* in_sizes, int n_in,
                              void* d_out, int out_size, void* d_ws, size_t ws_size,
                              hipStream_t stream) {
  // inputs (setup_inputs order)
  const int   *adj_rows = (const int*)d_in[0],  *adj_cols = (const int*)d_in[1];
  const float *adj_vals = (const float*)d_in[2];
  const int   *iir = (const int*)d_in[3], *iic = (const int*)d_in[4];
  const float *iiv = (const float*)d_in[5];
  const int   *itr = (const int*)d_in[6], *itc = (const int*)d_in[7];
  const float *itv = (const float*)d_in[8];
  const int   *rr  = (const int*)d_in[9], *rc  = (const int*)d_in[10];
  const float *rv  = (const float*)d_in[11];
  const float *user_emb = (const float*)d_in[12], *item_emb = (const float*)d_in[13];
  const float *image_feats = (const float*)d_in[14], *text_feats = (const float*)d_in[15];
  const float *img_res_w = (const float*)d_in[16];
  const float *img_res_g = (const float*)d_in[18], *img_res_beta = (const float*)d_in[19];
  const float *img_proj_w = (const float*)d_in[20];
  const float *img_proj_g = (const float*)d_in[22], *img_proj_beta = (const float*)d_in[23];
  const float *txt_res_w = (const float*)d_in[24];
  const float *txt_res_g = (const float*)d_in[26], *txt_res_beta = (const float*)d_in[27];
  const float *txt_proj_w = (const float*)d_in[28];
  const float *txt_proj_g = (const float*)d_in[30], *txt_proj_beta = (const float*)d_in[31];
  const float *gate_img_w = (const float*)d_in[32], *gate_img_b = (const float*)d_in[33];
  const float *gate_txt_w = (const float*)d_in[34], *gate_txt_b = (const float*)d_in[35];
  const float *common_w1 = (const float*)d_in[36], *common_b1 = (const float*)d_in[37];
  const float *common_w2 = (const float*)d_in[38];

  const int nnzUI = in_sizes[0], nnzII = in_sizes[3], nnzIT = in_sizes[6], nnzR = in_sizes[9];

  // workspace carve-out (fp32)
  float* F = (float*)d_ws;
  size_t o = 0;
  float* e0     = F + o; o += (size_t)NTOT * DDIM;   // later: content
  float* e1b    = F + o; o += (size_t)NTOT * DDIM;   // later: attention H
  float* e2b    = F + o; o += (size_t)NTOT * DDIM;
  float* img_ui = F + o; o += (size_t)NTOT * DDIM;
  float* txt_ui = F + o; o += (size_t)NTOT * DDIM;
  float* tY     = F + o; o += (size_t)NITEM * DDIM;  // raw GEMM out
  float* xImg   = F + o; o += (size_t)NITEM * DDIM;  // also spmm ping/pong
  float* fImg   = F + o; o += (size_t)NITEM * DDIM;
  float* xTxt   = F + o; o += (size_t)NITEM * DDIM;
  float* fTxt   = F + o; o += (size_t)NITEM * DDIM;
  float* itemA  = F + o; o += (size_t)NITEM * DDIM;
  float* stats  = F + o; o += 128;
  float* sImg   = F + o; o += NTOT;
  float* sTxt   = F + o; o += NTOT;
  o = (o + 63) & ~(size_t)63;
  bf16_t* Wb = (bf16_t*)(F + o);
  size_t ob = 0;
  bf16_t* WtImgRes  = Wb + ob; ob += (size_t)IMGF * DDIM;
  bf16_t* WtTxtRes  = Wb + ob; ob += (size_t)TXTF * DDIM;
  bf16_t* WtImgProj = Wb + ob; ob += DDIM * DDIM;
  bf16_t* WtTxtProj = Wb + ob; ob += DDIM * DDIM;
  bf16_t* WtGateImg = Wb + ob; ob += DDIM * DDIM;
  bf16_t* WtGateTxt = Wb + ob; ob += DDIM * DDIM;
  bf16_t* WtC1      = Wb + ob; ob += DDIM * DDIM;

  const int B = 256;
  const size_t bytesN = (size_t)NTOT * DDIM * sizeof(float);
  const size_t bytesI = (size_t)NITEM * DDIM * sizeof(float);

  // --- weight transposes to bf16 ---
  k_wt<<<cdiv((long)IMGF * DDIM, B), B, 0, stream>>>(img_res_w, WtImgRes, IMGF);
  k_wt<<<cdiv((long)TXTF * DDIM, B), B, 0, stream>>>(txt_res_w, WtTxtRes, TXTF);
  k_wt<<<cdiv(DDIM * DDIM, B), B, 0, stream>>>(img_proj_w, WtImgProj, DDIM);
  k_wt<<<cdiv(DDIM * DDIM, B), B, 0, stream>>>(txt_proj_w, WtTxtProj, DDIM);
  k_wt<<<cdiv(DDIM * DDIM, B), B, 0, stream>>>(gate_img_w, WtGateImg, DDIM);
  k_wt<<<cdiv(DDIM * DDIM, B), B, 0, stream>>>(gate_txt_w, WtGateTxt, DDIM);
  k_wt<<<cdiv(DDIM * DDIM, B), B, 0, stream>>>(common_w1, WtC1, DDIM);

  // --- user-item 2-layer GCN + mean ---
  k_concat<<<cdiv((long)NTOT * DDIM, B), B, 0, stream>>>(user_emb, item_emb, e0);
  (void)hipMemsetAsync(e1b, 0, bytesN, stream);
  k_spmm<<<cdiv((long)nnzUI * 16, B), B, 0, stream>>>(adj_rows, adj_cols, adj_vals, e0, e1b, nnzUI);
  (void)hipMemsetAsync(e2b, 0, bytesN, stream);
  k_spmm<<<cdiv((long)nnzUI * 16, B), B, 0, stream>>>(adj_rows, adj_cols, adj_vals, e1b, e2b, nnzUI);
  k_content<<<cdiv((long)NTOT * DDIM, B), B, 0, stream>>>(e0, e1b, e2b);  // e0 := content

  const int gI = cdiv(NITEM / 16, 4), gN = cdiv(cdiv(NTOT, 16), 4);

  // --- image branch: res-proj + proj (Linear+BN+Leaky, residual) ---
  k_gemm64<<<gI, 128, 0, stream>>>(image_feats, WtImgRes, tY, NITEM, IMGF);
  k_bnstats<<<64, 256, 0, stream>>>(tY, stats, NITEM);
  k_bnapply<<<cdiv((long)NITEM * DDIM, B), B, 0, stream>>>(tY, stats, img_res_g, img_res_beta, nullptr, xImg, NITEM);
  k_gemm64<<<gI, 128, 0, stream>>>(xImg, WtImgProj, tY, NITEM, DDIM);
  k_bnstats<<<64, 256, 0, stream>>>(tY, stats, NITEM);
  k_bnapply<<<cdiv((long)NITEM * DDIM, B), B, 0, stream>>>(tY, stats, img_proj_g, img_proj_beta, xImg, fImg, NITEM);

  // --- text branch ---
  k_gemm64<<<gI, 128, 0, stream>>>(text_feats, WtTxtRes, tY, NITEM, TXTF);
  k_bnstats<<<64, 256, 0, stream>>>(tY, stats, NITEM);
  k_bnapply<<<cdiv((long)NITEM * DDIM, B), B, 0, stream>>>(tY, stats, txt_res_g, txt_res_beta, nullptr, xTxt, NITEM);
  k_gemm64<<<gI, 128, 0, stream>>>(xTxt, WtTxtProj, tY, NITEM, DDIM);
  k_bnstats<<<64, 256, 0, stream>>>(tY, stats, NITEM);
  k_bnapply<<<cdiv((long)NITEM * DDIM, B), B, 0, stream>>>(tY, stats, txt_proj_g, txt_proj_beta, xTxt, fTxt, NITEM);

  // --- image item-item GCN with gating, then r-spmm to users ---
  k_gemm64<<<gI, 128, 0, stream>>>(fImg, WtGateImg, tY, NITEM, DDIM);
  k_gate<<<cdiv((long)NITEM * DDIM, B), B, 0, stream>>>(tY, gate_img_b, item_emb, itemA);
  (void)hipMemsetAsync(xImg, 0, bytesI, stream);
  k_spmm<<<cdiv((long)nnzII * 16, B), B, 0, stream>>>(iir, iic, iiv, itemA, xImg, nnzII);
  (void)hipMemsetAsync(img_ui, 0, bytesN, stream);
  k_spmm<<<cdiv((long)nnzII * 16, B), B, 0, stream>>>(iir, iic, iiv, xImg, img_ui + (size_t)NUSER * DDIM, nnzII);
  k_spmm<<<cdiv((long)nnzR * 16, B), B, 0, stream>>>(rr, rc, rv, img_ui + (size_t)NUSER * DDIM, img_ui, nnzR);

  // --- text item-item GCN with gating, then r-spmm to users ---
  k_gemm64<<<gI, 128, 0, stream>>>(fTxt, WtGateTxt, tY, NITEM, DDIM);
  k_gate<<<cdiv((long)NITEM * DDIM, B), B, 0, stream>>>(tY, gate_txt_b, item_emb, itemA);
  (void)hipMemsetAsync(xImg, 0, bytesI, stream);
  k_spmm<<<cdiv((long)nnzIT * 16, B), B, 0, stream>>>(itr, itc, itv, itemA, xImg, nnzIT);
  (void)hipMemsetAsync(txt_ui, 0, bytesN, stream);
  k_spmm<<<cdiv((long)nnzIT * 16, B), B, 0, stream>>>(itr, itc, itv, xImg, txt_ui + (size_t)NUSER * DDIM, nnzIT);
  k_spmm<<<cdiv((long)nnzR * 16, B), B, 0, stream>>>(rr, rc, rv, txt_ui + (size_t)NUSER * DDIM, txt_ui, nnzR);

  // --- common attention fusion ---
  k_gemm64<<<gN, 128, 0, stream>>>(img_ui, WtC1, e1b, NTOT, DDIM);
  k_score<<<cdiv(NTOT, B), B, 0, stream>>>(e1b, common_b1, common_w2, sImg, NTOT);
  k_gemm64<<<gN, 128, 0, stream>>>(txt_ui, WtC1, e1b, NTOT, DDIM);
  k_score<<<cdiv(NTOT, B), B, 0, stream>>>(e1b, common_b1, common_w2, sTxt, NTOT);

  k_fuse<<<cdiv((long)NTOT * DDIM, B), B, 0, stream>>>(e0, img_ui, txt_ui, sImg, sTxt, (float*)d_out);
}
